// GraphTransformer_75995151336010
// MI455X (gfx1250) — matrix operand order repaired and checked by reading, not compiled, not run on hardware
//
#include <hip/hip_runtime.h>
#include <hip/hip_bf16.h>

// ---------------------------------------------------------------------------
// Types for CDNA5 WMMA (wave32, 16x16x32 bf16 -> f32)
// ---------------------------------------------------------------------------
typedef __bf16 bf16_t;
typedef __attribute__((ext_vector_type(16))) __bf16 v16bf;
typedef __attribute__((ext_vector_type(8)))  __bf16 v8bf;
typedef __attribute__((ext_vector_type(4)))  __bf16 v4bf;
typedef __attribute__((ext_vector_type(8)))  float  v8f;

#define DEVINL __device__ __forceinline__

// ---------------------------------------------------------------------------
// Small utility kernels
// ---------------------------------------------------------------------------
__global__ __launch_bounds__(256) void fill_f32_kernel(float* p, float v, int n) {
    int t = blockIdx.x * blockDim.x + threadIdx.x;
    if (t < n) p[t] = v;
}

__global__ __launch_bounds__(256) void fill_i32_kernel(int* p, int v, int n) {
    int t = blockIdx.x * blockDim.x + threadIdx.x;
    if (t < n) p[t] = v;
}

__global__ __launch_bounds__(256) void cvt_f32_bf16_kernel(const float* __restrict__ s,
                                                           bf16_t* __restrict__ d, int n) {
    int t = blockIdx.x * blockDim.x + threadIdx.x;
    if (t < n) d[t] = (bf16_t)s[t];
}

__global__ __launch_bounds__(256) void copy_f32_kernel(const float* __restrict__ s,
                                                       float* __restrict__ d, int n) {
    int t = blockIdx.x * blockDim.x + threadIdx.x;
    if (t < n) d[t] = s[t];
}

// h = embed_w[x]  -> bf16 [N,128]
__global__ __launch_bounds__(256) void embed_nodes_kernel(const int* __restrict__ x,
                                                          const float* __restrict__ embw,
                                                          bf16_t* __restrict__ hbf, int total) {
    int t = blockIdx.x * blockDim.x + threadIdx.x;
    if (t >= total) return;
    int n = t >> 7, c = t & 127;
    hbf[t] = (bf16_t)embw[(size_t)x[n] * 128 + c];
}

// ea = embed_ee_w[edge_attr] -> bf16 [E,128]
__global__ __launch_bounds__(256) void embed_edges_kernel(const int* __restrict__ ea,
                                                          const float* __restrict__ embe,
                                                          bf16_t* __restrict__ eabf, int total) {
    int t = blockIdx.x * blockDim.x + threadIdx.x;
    if (t >= total) return;
    int e = t >> 7, c = t & 127;
    eabf[t] = (bf16_t)embe[(size_t)ea[e] * 128 + c];
}

// ---------------------------------------------------------------------------
// LDS-staged WMMA GEMM: Out[M,Nc] = A[M,K](bf16) * W[Nc,K]^T (+ bias)
//
// Block = 256 threads (8 waves). Each block stages a 128 x K activation slab
// into LDS ONCE via global_load_async_to_lds_b128 (ASYNCcnt), then wave w
// computes rows [16w,16w+16) for ALL column tiles, 4 tiles (16x64) at a time.
// Each A fragment (2 ds_load_b128) feeds 4 WMMAs. M%16==0, Nc%64==0, K%32==0.
// LDS rows padded by 8 bf16 (16B) so consecutive rows land 4 banks apart.
// ---------------------------------------------------------------------------
template <typename OutT>
__global__ __launch_bounds__(256) void gemm_lds_wmma_kernel(
    const bf16_t* __restrict__ A, const bf16_t* __restrict__ W,
    const float* __restrict__ bias, OutT* __restrict__ Out,
    int M, int Nc, int K) {
    extern __shared__ bf16_t lds_a[];  // 128 * (K+8) bf16, passed at launch

    const int wave = threadIdx.x >> 5;
    const int lane = threadIdx.x & 31;
    const int lhi  = lane >> 4;   // 0 or 1
    const int llo  = lane & 15;
    const int stride  = K + 8;           // padded LDS row stride (elements)
    const int kchunks = K >> 3;          // 16B chunks per row (power of 2)
    const int kclog   = 31 - __clz(kchunks);
    const int rowBlk  = blockIdx.x * 128;

    // ---- stage A[rowBlk : rowBlk+128, 0:K] into LDS (async, no VGPR data) ----
    const int iters = (kchunks >> 1);    // (16 rows * kchunks) / 32 lanes
    for (int i = 0; i < iters; ++i) {
        int f    = lane + (i << 5);
        int r    = (wave << 4) + (f >> kclog);
        int kc   = (f & (kchunks - 1)) << 3;           // element offset in row
        int grow = rowBlk + r;
        if (grow > M - 1) grow = M - 1;                // clamp (tail block)
        const bf16_t* gsrc = A + (size_t)grow * K + kc;
        uint32_t loff = (uint32_t)(uintptr_t)(&lds_a[r * stride + kc]);
        asm volatile("global_load_async_to_lds_b128 %0, %1, off"
                     :: "v"(loff), "v"(gsrc) : "memory");
    }
    asm volatile("s_wait_asynccnt 0x0" ::: "memory");
    __syncthreads();

    // ---- compute: wave-uniform activity check keeps EXEC all-ones for WMMA ----
    const int tileRow = blockIdx.x * 8 + wave;
    if (tileRow < (M >> 4)) {
        const bf16_t* aL = lds_a + (size_t)(wave * 16 + llo) * stride;
        const int nGroups = Nc >> 6;                   // groups of 4 col-tiles
        for (int cg = 0; cg < nGroups; ++cg) {
            const int ncol0 = cg * 64;
            v8f acc[4];
            const bf16_t* wr[4];
#pragma unroll
            for (int t = 0; t < 4; ++t) {
                float b0 = bias ? bias[ncol0 + t * 16 + llo] : 0.0f;
#pragma unroll
                for (int r = 0; r < 8; ++r) acc[t][r] = b0;
                wr[t] = W + (size_t)(ncol0 + t * 16 + llo) * K;
            }
            for (int k0 = 0; k0 < K; k0 += 32) {
                // A 16x32 fragment from LDS (ds_load_b128 x2)
                v8bf alo = *(const v8bf*)(aL + k0 + 8 * lhi);
                v8bf ahi = *(const v8bf*)(aL + k0 + 16 + 8 * lhi);
                v16bf a = __builtin_shufflevector(alo, ahi,
                    0, 1, 2, 3, 4, 5, 6, 7, 8, 9, 10, 11, 12, 13, 14, 15);
#pragma unroll
                for (int t = 0; t < 4; ++t) {
                    v8bf blo = *(const v8bf*)(wr[t] + k0 + 8 * lhi);
                    v8bf bhi = *(const v8bf*)(wr[t] + k0 + 16 + 8 * lhi);
                    v16bf b = __builtin_shufflevector(blo, bhi,
                        0, 1, 2, 3, 4, 5, 6, 7, 8, 9, 10, 11, 12, 13, 14, 15);
                    acc[t] = __builtin_amdgcn_wmma_f32_16x16x32_bf16(
                        false, a, false, b, (short)0, acc[t], false, false);
                }
            }
            // C/D: lane n = llo ; VGPR r -> m = r + 8*lhi
#pragma unroll
            for (int t = 0; t < 4; ++t) {
#pragma unroll
                for (int r = 0; r < 8; ++r) {
                    int m = tileRow * 16 + r + 8 * lhi;
                    Out[(size_t)m * Nc + ncol0 + t * 16 + llo] = (OutT)acc[t][r];
                }
            }
        }
    }
}

// ---------------------------------------------------------------------------
// Attention kernels (H=2 heads, C=128). QKVS fused: row stride `st`,
// Q at col 0, K at 256, V at 512, S(skip) at 768.
// ---------------------------------------------------------------------------
DEVINL int f32_to_ordered(float f) {
    int i = __float_as_int(f);
    return i >= 0 ? i : (i ^ 0x7fffffff);
}
DEVINL float ordered_to_f32(int k) {
    int i = k >= 0 ? k : (k ^ 0x7fffffff);
    return __int_as_float(i);
}

// One 32-lane wave per (edge, head): alpha = <q[dst], k[src]+epj> / sqrt(128)
__global__ __launch_bounds__(256) void attn_alpha_kernel(
    const float* __restrict__ QKVS, int st, const bf16_t* __restrict__ Epj,
    const int* __restrict__ src, const int* __restrict__ dst,
    float* __restrict__ alpha, int* __restrict__ amax_i, int E2) {
    int wave = threadIdx.x >> 5, lane = threadIdx.x & 31;
    int p = blockIdx.x * 8 + wave;
    if (p >= E2) return;
    int e = p >> 1, hh = p & 1;
    int s = src[e], d = dst[e];
    const float4 q4 = *(const float4*)(QKVS + (size_t)d * st + hh * 128 + lane * 4);
    const float4 k4 = *(const float4*)(QKVS + (size_t)s * st + 256 + hh * 128 + lane * 4);
    const v4bf  e4 = *(const v4bf*)(Epj + (size_t)e * 256 + hh * 128 + lane * 4);
    float sum = q4.x * (k4.x + (float)e4[0]) + q4.y * (k4.y + (float)e4[1]) +
                q4.z * (k4.z + (float)e4[2]) + q4.w * (k4.w + (float)e4[3]);
#pragma unroll
    for (int off = 16; off > 0; off >>= 1) sum += __shfl_xor(sum, off, 32);
    if (lane == 0) {
        float a = sum * 0.08838834764831845f;  // 1/sqrt(128)
        alpha[p] = a;
        atomicMax(&amax_i[d * 2 + hh], f32_to_ordered(a));
    }
}

// ex = exp(alpha - amax[dst]); denom[dst] += ex  (alpha overwritten with ex)
__global__ __launch_bounds__(256) void attn_exp_kernel(
    const int* __restrict__ dst, const int* __restrict__ amax_i,
    float* __restrict__ alpha, float* __restrict__ denom, int E2) {
    int p = blockIdx.x * blockDim.x + threadIdx.x;
    if (p >= E2) return;
    int e = p >> 1, hh = p & 1;
    int d = dst[e];
    float m = ordered_to_f32(amax_i[d * 2 + hh]);
    float ex = __expf(alpha[p] - m);
    alpha[p] = ex;
    atomicAdd(&denom[d * 2 + hh], ex);
}

// acc[dst, i] += (v[src, i] + epj[e, i]) * ex/(denom[dst]+eps)
__global__ __launch_bounds__(256) void attn_msg_kernel(
    const float* __restrict__ QKVS, int st, const bf16_t* __restrict__ Epj,
    const float* __restrict__ alpha, const float* __restrict__ denom,
    const int* __restrict__ src, const int* __restrict__ dst,
    float* __restrict__ acc, int total /* E*256 */) {
    int t = blockIdx.x * blockDim.x + threadIdx.x;
    if (t >= total) return;
    int e = t >> 8, i = t & 255, hh = i >> 7;
    int s = src[e], d = dst[e];
    float w = alpha[e * 2 + hh] / (denom[d * 2 + hh] + 1e-16f);
    float val = (QKVS[(size_t)s * st + 512 + i] + (float)Epj[(size_t)e * 256 + i]) * w;
    atomicAdd(&acc[(size_t)d * 256 + i], val);
}

// concat layers: h = gelu(acc + skip) -> bf16 for next layer's GEMMs
__global__ __launch_bounds__(256) void combine_concat_kernel(
    const float* __restrict__ acc, const float* __restrict__ QKVS, int st,
    bf16_t* __restrict__ hbf, int total /* N*256 */) {
    int t = blockIdx.x * blockDim.x + threadIdx.x;
    if (t >= total) return;
    int n = t >> 8, i = t & 255;
    float x = acc[t] + QKVS[(size_t)n * st + 768 + i];
    float g = 0.5f * x * (1.0f + erff(x * 0.7071067811865476f));  // exact GELU
    hbf[t] = (bf16_t)g;
}

// final layer: out = mean over heads + skip; only readout nodes 0..G-1 needed
__global__ __launch_bounds__(256) void combine_mean_kernel(
    const float* __restrict__ acc, const float* __restrict__ QKVS, int st,
    float* __restrict__ out, int total /* G*128 */) {
    int t = blockIdx.x * blockDim.x + threadIdx.x;
    if (t >= total) return;
    int n = t >> 7, c = t & 127;
    float x = 0.5f * (acc[(size_t)n * 256 + c] + acc[(size_t)n * 256 + 128 + c]) +
              QKVS[(size_t)n * st + 768 + c];
    out[t] = x;
}

// ---------------------------------------------------------------------------
// Host orchestration
// ---------------------------------------------------------------------------
static inline int cdiv(int a, int b) { return (a + b - 1) / b; }

extern "C" void kernel_launch(void* const* d_in, const int* in_sizes, int n_in,
                              void* d_out, int out_size, void* d_ws, size_t ws_size,
                              hipStream_t stream) {
    constexpr int N = 50000, E = 400000, C = 128, G = 1024;
    constexpr int HC = 256;  // H * C

    const int*   x     = (const int*)d_in[0];
    const int*   eidx  = (const int*)d_in[1];
    const int*   src   = eidx;
    const int*   dstp  = eidx + E;
    const int*   eattr = (const int*)d_in[3];
    const float* embw  = (const float*)d_in[5];
    const float* embe  = (const float*)d_in[6];

    // Workspace carve (256B aligned)
    char* wp = (char*)d_ws;
    auto carve = [&](size_t bytes) -> void* {
        void* r = (void*)wp;
        wp += (bytes + 255) & ~(size_t)255;
        return r;
    };
    bf16_t* hbf    = (bf16_t*)carve((size_t)N * HC * 2);    // activations (bf16)
    bf16_t* eabf   = (bf16_t*)carve((size_t)E * C * 2);     // edge embeddings (bf16)
    bf16_t* epjbf  = (bf16_t*)carve((size_t)E * HC * 2);    // projected edge feats
    float*  qkvs   = (float*)carve((size_t)N * 1024 * 4);   // fused [Q|K|V|S]
    float*  accf   = (float*)carve((size_t)N * HC * 4);     // message accumulator
    float*  alpha  = (float*)carve((size_t)E * 2 * 4);
    int*    amax_i = (int*)carve((size_t)N * 2 * 4);
    float*  denom  = (float*)carve((size_t)N * 2 * 4);
    bf16_t* wall   = (bf16_t*)carve((size_t)1024 * HC * 2); // fused bf16 weights
    float*  ball   = (float*)carve((size_t)1024 * 4);       // fused bias
    bf16_t* webf   = (bf16_t*)carve((size_t)HC * C * 2);    // edge-proj weights
    (void)ws_size; (void)n_in; (void)in_sizes; (void)out_size;

    const int B = 256;

    // Embedding gathers
    embed_nodes_kernel<<<cdiv(N * C, B), B, 0, stream>>>(x, embw, hbf, N * C);
    embed_edges_kernel<<<cdiv(E * C, B), B, 0, stream>>>(eattr, embe, eabf, E * C);

    for (int layer = 0; layer < 3; ++layer) {
        const int Ki   = (layer == 0) ? C : HC;   // input dim
        const int Oi   = (layer == 2) ? C : HC;   // skip/out dim
        const int Ntot = 768 + Oi;                // fused output width
        const int pb   = 7 + layer * 9;           // params in setup_inputs() dict order
        const float* Wq = (const float*)d_in[pb + 0];
        const float* bq = (const float*)d_in[pb + 1];
        const float* Wk = (const float*)d_in[pb + 2];
        const float* bk = (const float*)d_in[pb + 3];
        const float* Wv = (const float*)d_in[pb + 4];
        const float* bv = (const float*)d_in[pb + 5];
        const float* We = (const float*)d_in[pb + 6];
        const float* Ws = (const float*)d_in[pb + 7];
        const float* bs = (const float*)d_in[pb + 8];

        // Stage fused weights [Wq;Wk;Wv;Ws] + bias in bf16/f32 (L2 resident)
        cvt_f32_bf16_kernel<<<cdiv(HC * Ki, B), B, 0, stream>>>(Wq, wall + (size_t)0 * Ki, HC * Ki);
        cvt_f32_bf16_kernel<<<cdiv(HC * Ki, B), B, 0, stream>>>(Wk, wall + (size_t)256 * Ki, HC * Ki);
        cvt_f32_bf16_kernel<<<cdiv(HC * Ki, B), B, 0, stream>>>(Wv, wall + (size_t)512 * Ki, HC * Ki);
        cvt_f32_bf16_kernel<<<cdiv(Oi * Ki, B), B, 0, stream>>>(Ws, wall + (size_t)768 * Ki, Oi * Ki);
        copy_f32_kernel<<<1, B, 0, stream>>>(bq, ball + 0, HC);
        copy_f32_kernel<<<1, B, 0, stream>>>(bk, ball + 256, HC);
        copy_f32_kernel<<<1, B, 0, stream>>>(bv, ball + 512, HC);
        copy_f32_kernel<<<1, B, 0, stream>>>(bs, ball + 768, Oi);
        cvt_f32_bf16_kernel<<<cdiv(HC * C, B), B, 0, stream>>>(We, webf, HC * C);

        // Fused Q/K/V/S projection: one LDS-staged pass over activations
        gemm_lds_wmma_kernel<float>
            <<<cdiv(N, 128), B, (size_t)128 * (Ki + 8) * 2, stream>>>(
                hbf, wall, ball, qkvs, N, Ntot, Ki);
        // Edge projection (bf16 out, no bias)
        gemm_lds_wmma_kernel<bf16_t>
            <<<cdiv(E, 128), B, (size_t)128 * (C + 8) * 2, stream>>>(
                eabf, webf, nullptr, epjbf, E, HC, C);

        // Attention: segment softmax over incoming edges per (dst, head)
        fill_i32_kernel<<<cdiv(N * 2, B), B, 0, stream>>>(amax_i, INT_MIN, N * 2);
        fill_f32_kernel<<<cdiv(N * 2, B), B, 0, stream>>>(denom, 0.0f, N * 2);
        fill_f32_kernel<<<cdiv(N * HC, B), B, 0, stream>>>(accf, 0.0f, N * HC);

        attn_alpha_kernel<<<cdiv(E * 2, 8), B, 0, stream>>>(
            qkvs, Ntot, epjbf, src, dstp, alpha, amax_i, E * 2);
        attn_exp_kernel<<<cdiv(E * 2, B), B, 0, stream>>>(
            dstp, amax_i, alpha, denom, E * 2);
        attn_msg_kernel<<<cdiv(E * HC, B), B, 0, stream>>>(
            qkvs, Ntot, epjbf, alpha, denom, src, dstp, accf, E * HC);

        if (layer < 2) {
            combine_concat_kernel<<<cdiv(N * HC, B), B, 0, stream>>>(
                accf, qkvs, Ntot, hbf, N * HC);
        } else {
            // readout nodes are exactly the first G nodes (token id 0)
            combine_mean_kernel<<<cdiv(G * C, B), B, 0, stream>>>(
                accf, qkvs, Ntot, (float*)d_out, G * C);
        }
    }
}
